// GCPATr_79456894976576
// MI455X (gfx1250) — compile-verified, matrix operand order checked by reading, not run on hardware
//
#include <hip/hip_runtime.h>
#include <hip/hip_bf16.h>
#include <cstdint>

#define BB 8
#define TT 1024
#define NN 64
#define HH 8
#define LL 2
#define KKS 3
#define DD1 64
#define EPSF 1e-6f

typedef __attribute__((ext_vector_type(16))) __bf16 v16bf;
typedef __attribute__((ext_vector_type(8)))  float  v8f;
typedef __attribute__((ext_vector_type(8)))  int    v8i_;
typedef unsigned short u16;

#define WMMA_BF16(A, Bm, C) \
  __builtin_amdgcn_wmma_f32_16x16x32_bf16(false, (A), false, (Bm), (short)0, (C), false, false)

__device__ __forceinline__ u16 f2bf(float f) {
  unsigned u = __float_as_uint(f);
  u += 0x7FFFu + ((u >> 16) & 1u);
  return (u16)(u >> 16);
}
__device__ __forceinline__ float bf2f(u16 b) {
  return __uint_as_float(((unsigned)b) << 16);
}
__device__ __forceinline__ v8f vzero() {
  v8f v;
#pragma unroll
  for (int i = 0; i < 8; ++i) v[i] = 0.f;
  return v;
}

// ---- WMMA bf16 fragment loaders: 8 aligned dword loads, no 16-bit packing ----
// Fragment element pairs (2p,2p+1) sit at K = ((p>>2)<<4) + koff + 2*(p&3), K+1.
// A fragment from row-major A (MxK), leading dim lda: lane's K-run contiguous.
__device__ __forceinline__ v16bf load_a_rm(const u16* src, int lda) {
  int lane = threadIdx.x & 31;
  int row  = lane & 15;
  int koff = (lane >> 4) << 3;
  const u16* r = src + row * lda + koff;
  v8i_ iv;
#pragma unroll
  for (int p = 0; p < 8; ++p) {
    int k = ((p >> 2) << 4) + ((p & 3) << 1);
    iv[p] = *(const int*)(r + k);
  }
  return __builtin_bit_cast(v16bf, iv);
}
// B fragment where logical B[k][col] = src[col*ld + k] (source holds B^T rows,
// i.e. source is [N][K]); lane's K-run contiguous.
__device__ __forceinline__ v16bf load_b_tr(const u16* src, int ld) {
  int lane = threadIdx.x & 31;
  int col  = lane & 15;
  int koff = (lane >> 4) << 3;
  const u16* r = src + col * ld + koff;
  v8i_ iv;
#pragma unroll
  for (int p = 0; p < 8; ++p) {
    int k = ((p >> 2) << 4) + ((p & 3) << 1);
    iv[p] = *(const int*)(r + k);
  }
  return __builtin_bit_cast(v16bf, iv);
}

// ============================== prep kernels ===============================

__global__ void k_addpos(const float* __restrict__ in, const float* __restrict__ pe,
                         float* __restrict__ x, int n) {
  for (int i = blockIdx.x * blockDim.x + threadIdx.x; i < n; i += gridDim.x * blockDim.x)
    x[i] = in[i] + pe[i & (TT * NN - 1)];
}

// batched transpose-convert: src[b][r][c] f32 -> dst[b][c][r] bf16  ([N][K] B layout)
__global__ void k_f2bf_tr(const float* __restrict__ src, u16* __restrict__ dst,
                          int rows, int cols, int total) {
  for (int i = blockIdx.x * blockDim.x + threadIdx.x; i < total; i += gridDim.x * blockDim.x) {
    int c = i % cols;
    int r = (i / cols) % rows;
    int b = i / (rows * cols);
    dst[(size_t)b * rows * cols + (size_t)c * rows + r] = f2bf(src[i]);
  }
}

// pack qb/kb/vb -> bias3h[l][w][h]
__global__ void k_pack_bias(const float* __restrict__ qb, const float* __restrict__ kb,
                            const float* __restrict__ vb, float* __restrict__ dst) {
  int i = threadIdx.x;
  if (i < LL * 3 * HH) {
    int h = i % HH;
    int w = (i / HH) % 3;
    int l = i / (3 * HH);
    const float* src = (w == 0 ? qb : (w == 1 ? kb : vb));
    dst[i] = src[l * HH + h];
  }
}

// out[l][w][h][m][n] = sum_k ws[l,h,k]*K3[n,k,m] + (n==m)  (transposed store: [N][K])
__global__ void k_build_mqkv(const float* __restrict__ K3, const float* __restrict__ qw,
                             const float* __restrict__ kw, const float* __restrict__ vw,
                             u16* __restrict__ out) {
  int total = LL * 3 * HH * NN * NN;
  for (int i = blockIdx.x * blockDim.x + threadIdx.x; i < total; i += gridDim.x * blockDim.x) {
    int m = i % NN;
    int n = (i / NN) % NN;
    int h = (i / (NN * NN)) % HH;
    int w = (i / (NN * NN * HH)) % 3;
    int l = i / (NN * NN * HH * 3);
    const float* ws = (w == 0 ? qw : (w == 1 ? kw : vw)) + (l * HH + h) * KKS;
    float acc = (n == m) ? 1.f : 0.f;
    for (int k = 0; k < KKS; ++k) acc += ws[k] * K3[(n * KKS + k) * NN + m];
    size_t base = (size_t)(((l * 3 + w) * HH) + h) * NN * NN;
    out[base + (size_t)m * NN + n] = f2bf(acc);   // transposed: [out col m][in row n]
  }
}

// out[l][m][n] = sum_k sclw[l,k]*K3[n,k,m] + (n==m)   (transposed store)
__global__ void k_build_msc(const float* __restrict__ K3, const float* __restrict__ sclw,
                            u16* __restrict__ out) {
  int total = LL * NN * NN;
  for (int i = blockIdx.x * blockDim.x + threadIdx.x; i < total; i += gridDim.x * blockDim.x) {
    int m = i % NN;
    int n = (i / NN) % NN;
    int l = i / (NN * NN);
    float acc = (n == m) ? 1.f : 0.f;
    for (int k = 0; k < KKS; ++k) acc += sclw[l * KKS + k] * K3[(n * KKS + k) * NN + m];
    out[(size_t)l * NN * NN + (size_t)m * NN + n] = f2bf(acc);
  }
}

// ======================= QKV graph-conv projections ========================
// xqkv layout: [w(3)][B][H][T][N] bf16; out pointer is pure arithmetic (no branches).
#define QKV_SZ ((size_t)BB * HH * TT * NN)
__global__ void k_qkv(const float* __restrict__ x, const u16* __restrict__ mqkv,
                      const float* __restrict__ bias3h, u16* __restrict__ xqkv) {
  int r0 = blockIdx.x * 64;
  __shared__ __align__(16) u16 xb[64 * 64];
  int tid = threadIdx.x, wid = tid >> 5;
  {
    const float2* xs = (const float2*)(x + (size_t)r0 * 64);
    unsigned* xd = (unsigned*)xb;
    for (int idx = tid; idx < 2048; idx += 256) {
      float2 v = xs[idx];
      xd[idx] = (unsigned)f2bf(v.x) | ((unsigned)f2bf(v.y) << 16);
    }
  }
  __syncthreads();
  int lane = tid & 31, ncol = lane & 15, moff = (lane >> 4) << 3;
  for (int j = wid; j < 384; j += 8) {
    int w = j >> 7;         // 0..2 : q/k/v
    int rem = j & 127;
    int h = rem >> 4;
    int tile = rem & 15;
    int tm = tile >> 2, tn = tile & 3;
    const u16* M = mqkv + (size_t)(w * HH + h) * 4096;   // [N=m][K=n] transposed
    v8f acc = vzero();
#pragma unroll
    for (int kkk = 0; kkk < 64; kkk += 32) {
      v16bf a  = load_a_rm(xb + tm * 16 * 64 + kkk, 64);
      v16bf bm = load_b_tr(M + (size_t)(tn * 16) * 64 + kkk, 64);
      acc = WMMA_BF16(a, bm, acc);
    }
    float bias = bias3h[w * HH + h];
    u16* out = xqkv + (size_t)w * QKV_SZ;
#pragma unroll
    for (int v = 0; v < 8; ++v) {
      int m = v + moff;
      int g = r0 + tm * 16 + m;
      int b_ = g >> 10, t = g & 1023;
      float val = fmaxf(acc[v] + bias, 0.f);
      out[(((size_t)b_ * HH + h) * TT + t) * NN + tn * 16 + ncol] = f2bf(val);
    }
  }
}

// ============== scores S = Xk Xq^T / N, row softmax -> sft (bf16) ==========
__global__ void k_scores(const u16* __restrict__ xq, const u16* __restrict__ xk,
                         u16* __restrict__ sft) {
  extern __shared__ char smem[];
  float* S = (float*)smem;                                      // 64*1024 f32
  u16* kt  = (u16*)(smem + (size_t)64 * 1024 * sizeof(float));  // 64*64 bf16
  int bh = blockIdx.y;
  int t0 = blockIdx.x * 64;
  const u16* xqb = xq + (size_t)bh * TT * NN;
  const u16* xkb = xk + (size_t)bh * TT * NN;
  int tid = threadIdx.x, wid = tid >> 5;
  {
    const unsigned* ksrc = (const unsigned*)(xkb + (size_t)t0 * 64);
    unsigned* kdst = (unsigned*)kt;
    for (int idx = tid; idx < 2048; idx += 256) kdst[idx] = ksrc[idx];
  }
  __syncthreads();
  int lane = tid & 31, ncol = lane & 15, moff = (lane >> 4) << 3;
  for (int j = wid; j < 256; j += 8) {
    int tm = j >> 6;      // t tile 0..3
    int sn = j & 63;      // s tile 0..63
    v8f acc = vzero();
#pragma unroll
    for (int kkk = 0; kkk < 64; kkk += 32) {
      v16bf a  = load_a_rm(kt + tm * 16 * 64 + kkk, 64);
      v16bf bm = load_b_tr(xqb + (size_t)(sn * 16) * 64 + kkk, 64);  // B[k][c]=xq[s0+c][k]
      acc = WMMA_BF16(a, bm, acc);
    }
#pragma unroll
    for (int v = 0; v < 8; ++v) {
      int m = v + moff;
      S[(tm * 16 + m) * 1024 + sn * 16 + ncol] = acc[v] * (1.f / 64.f);
    }
  }
  __syncthreads();
  // row softmax: 4 lanes per row, reduce with shfl_xor within wave32
  int row = tid >> 2, q = tid & 3;
  float* Sr = S + row * 1024;
  float mx = -3.4e38f;
  for (int c = q * 256; c < q * 256 + 256; ++c) mx = fmaxf(mx, Sr[c]);
  mx = fmaxf(mx, __shfl_xor(mx, 1, 32));
  mx = fmaxf(mx, __shfl_xor(mx, 2, 32));
  float sum = 0.f;
  for (int c = q * 256; c < q * 256 + 256; ++c) {
    float e = __expf(Sr[c] - mx);
    Sr[c] = e;
    sum += e;
  }
  sum += __shfl_xor(sum, 1, 32);
  sum += __shfl_xor(sum, 2, 32);
  float inv = 1.f / sum;
  u16* dst = sft + ((size_t)bh * TT + t0 + row) * TT;
  for (int c = q * 256; c < q * 256 + 256; c += 2) {
    unsigned lo = f2bf(Sr[c] * inv);
    unsigned hi = f2bf(Sr[c + 1] * inv);
    *(unsigned*)(dst + c) = lo | (hi << 16);
  }
}

// ======================= auv = sft @ xv  (f32 out) =========================
__global__ void k_auv(const u16* __restrict__ sft, const u16* __restrict__ xv,
                      float* __restrict__ auv) {
  int bh = blockIdx.y;
  int t0 = blockIdx.x * 64;
  const u16* sfb = sft + (size_t)bh * TT * TT;
  const u16* xvb = xv + (size_t)bh * TT * NN;
  __shared__ __align__(16) u16 xvt[64 * 32];   // transposed tile: [n][k_local]
  int tid = threadIdx.x, wid = tid >> 5, j0 = wid * 2;
  v8f acc[2];
  acc[0] = vzero();
  acc[1] = vzero();
  for (int kk = 0; kk < TT; kk += 32) {
    for (int idx = tid; idx < 2048; idx += 256) {
      int i = idx >> 6, c = idx & 63;                 // i = k_local, c = n
      xvt[c * 32 + i] = xvb[(size_t)(kk + i) * 64 + c];
    }
    __syncthreads();
#pragma unroll
    for (int j2 = 0; j2 < 2; ++j2) {
      int j = j0 + j2;
      int tm = j >> 2, tn = j & 3;
      v16bf a  = load_a_rm(sfb + (size_t)(t0 + tm * 16) * TT + kk, TT);
      v16bf bm = load_b_tr(xvt + tn * 16 * 32, 32);
      acc[j2] = WMMA_BF16(a, bm, acc[j2]);
    }
    __syncthreads();
  }
  int lane = tid & 31, n = lane & 15, moff = (lane >> 4) << 3;
#pragma unroll
  for (int j2 = 0; j2 < 2; ++j2) {
    int j = j0 + j2;
    int tm = j >> 2, tn = j & 3;
#pragma unroll
    for (int v = 0; v < 8; ++v) {
      int m = v + moff;
      auv[((size_t)bh * TT + t0 + tm * 16 + m) * NN + tn * 16 + n] = acc[j2][v];
    }
  }
}

// ===== gate: h1=relu(sft^T@W1+b1), h2=relu(h1@W2+b2), g=softmax(h2@W3+b3) ==
// w1B is transposed [D1][T] per (l,h); w2B is transposed [e][d] per (l,h).
__global__ void k_gate(const u16* __restrict__ sft, const u16* __restrict__ w1B,
                       const float* __restrict__ tqb1, const u16* __restrict__ w2B,
                       const float* __restrict__ tqb2, const float* __restrict__ tqW3,
                       const float* __restrict__ tqb3, float* __restrict__ gate) {
  int bh = blockIdx.y;
  int h = bh & (HH - 1);
  int t0 = blockIdx.x * 64;
  const u16* sfb = sft + (size_t)bh * TT * TT;
  const u16* w1 = w1B + (size_t)h * TT * DD1;   // [d][s]
  const u16* w2 = w2B + (size_t)h * DD1 * DD1;  // [e][d]
  __shared__ __align__(16) u16 stT[64 * 32];    // A tile row-major: [t_local][k_local]
  __shared__ __align__(16) u16 h1b[64 * 64];
  __shared__ float h2f[64 * 64];
  int tid = threadIdx.x, wid = tid >> 5, j0 = wid * 2;
  int lane = tid & 31, n = lane & 15, moff = (lane >> 4) << 3;
  v8f acc[2];
  acc[0] = vzero();
  acc[1] = vzero();
  for (int kk = 0; kk < TT; kk += 32) {
    for (int idx = tid; idx < 2048; idx += 256) {
      int i = idx >> 6, c = idx & 63;             // i = k_local (s), c = t_local
      stT[c * 32 + i] = sfb[(size_t)(kk + i) * TT + t0 + c];  // A[t][s] = sft_mem[s][t]
    }
    __syncthreads();
#pragma unroll
    for (int j2 = 0; j2 < 2; ++j2) {
      int j = j0 + j2;
      int tm = j >> 2, tn = j & 3;
      v16bf a  = load_a_rm(stT + tm * 16 * 32, 32);
      v16bf bm = load_b_tr(w1 + (size_t)(tn * 16) * TT + kk, TT);
      acc[j2] = WMMA_BF16(a, bm, acc[j2]);
    }
    __syncthreads();
  }
#pragma unroll
  for (int j2 = 0; j2 < 2; ++j2) {
    int j = j0 + j2;
    int tm = j >> 2, tn = j & 3;
#pragma unroll
    for (int v = 0; v < 8; ++v) {
      int m = v + moff;
      float val = fmaxf(acc[j2][v] + tqb1[h * DD1 + tn * 16 + n], 0.f);
      h1b[(tm * 16 + m) * 64 + tn * 16 + n] = f2bf(val);
    }
  }
  __syncthreads();
  v8f acc2[2];
  acc2[0] = vzero();
  acc2[1] = vzero();
#pragma unroll
  for (int kkk = 0; kkk < 64; kkk += 32) {
#pragma unroll
    for (int j2 = 0; j2 < 2; ++j2) {
      int j = j0 + j2;
      int tm = j >> 2, tn = j & 3;
      v16bf a  = load_a_rm(h1b + tm * 16 * 64 + kkk, 64);
      v16bf bm = load_b_tr(w2 + (size_t)(tn * 16) * 64 + kkk, 64);
      acc2[j2] = WMMA_BF16(a, bm, acc2[j2]);
    }
  }
#pragma unroll
  for (int j2 = 0; j2 < 2; ++j2) {
    int j = j0 + j2;
    int tm = j >> 2, tn = j & 3;
#pragma unroll
    for (int v = 0; v < 8; ++v) {
      int m = v + moff;
      h2f[(tm * 16 + m) * 64 + tn * 16 + n] =
          fmaxf(acc2[j2][v] + tqb2[h * DD1 + tn * 16 + n], 0.f);
    }
  }
  __syncthreads();
  if (tid < 64) {
    int row = tid;
    const float* w3 = tqW3 + (size_t)h * DD1 * 2;
    float g0 = tqb3[h * 2 + 0], g1 = tqb3[h * 2 + 1];
    for (int d = 0; d < 64; ++d) {
      float v = h2f[row * 64 + d];
      g0 += v * w3[d * 2];
      g1 += v * w3[d * 2 + 1];
    }
    float m = fmaxf(g0, g1);
    float e0 = __expf(g0 - m), e1 = __expf(g1 - m);
    float inv = 1.f / (e0 + e1);
    size_t gi = ((size_t)bh * TT + t0 + row) * 2;
    gate[gi] = e0 * inv;
    gate[gi + 1] = e1 * inv;
  }
}

// ========= ho = g0*auv + g1*xv, concat heads -> a_x[B,T,H*N] (bf16) ========
__global__ void k_mix(const float* __restrict__ gate, const float* __restrict__ auv,
                      const u16* __restrict__ xv, u16* __restrict__ a_x) {
  int total = BB * HH * TT * NN;
  for (int i = blockIdx.x * blockDim.x + threadIdx.x; i < total; i += gridDim.x * blockDim.x) {
    int n = i & 63;
    int t = (i >> 6) & 1023;
    int bh = i >> 16;
    int b = bh >> 3, h = bh & 7;
    float g0 = gate[(size_t)(i >> 6) * 2];
    float g1 = gate[(size_t)(i >> 6) * 2 + 1];
    float val = g0 * auv[i] + g1 * bf2f(xv[i]);
    a_x[((size_t)(b * TT + t)) * (HH * NN) + h * 64 + n] = f2bf(val);
  }
}

// ========== out1 = LN(x + a_x @ Wlt) : K=512 GEMM + fused residual/LN ======
// wltB is transposed [64][512].
__global__ void k_proj_ln1(const float* __restrict__ x, const u16* __restrict__ a_x,
                           const u16* __restrict__ wltB, const float* __restrict__ lg,
                           const float* __restrict__ lb, float* __restrict__ out1f,
                           u16* __restrict__ out1b) {
  int r0 = blockIdx.x * 64;
  __shared__ float y[64 * 64];
  int tid = threadIdx.x, wid = tid >> 5, j0 = wid * 2;
  v8f acc[2];
  acc[0] = vzero();
  acc[1] = vzero();
#pragma unroll 2
  for (int kk = 0; kk < 512; kk += 32) {
#pragma unroll
    for (int j2 = 0; j2 < 2; ++j2) {
      int j = j0 + j2;
      int tm = j >> 2, tn = j & 3;
      v16bf a  = load_a_rm(a_x + (size_t)(r0 + tm * 16) * 512 + kk, 512);
      v16bf bm = load_b_tr(wltB + (size_t)(tn * 16) * 512 + kk, 512);
      acc[j2] = WMMA_BF16(a, bm, acc[j2]);
    }
  }
  int lane = tid & 31, n = lane & 15, moff = (lane >> 4) << 3;
#pragma unroll
  for (int j2 = 0; j2 < 2; ++j2) {
    int j = j0 + j2;
    int tm = j >> 2, tn = j & 3;
#pragma unroll
    for (int v = 0; v < 8; ++v) {
      int m = v + moff;
      y[(tm * 16 + m) * 64 + tn * 16 + n] =
          acc[j2][v] + x[(size_t)(r0 + tm * 16 + m) * 64 + tn * 16 + n];
    }
  }
  __syncthreads();
  if (tid < 64) {
    int row = tid;
    float mean = 0.f, var = 0.f;
    for (int c = 0; c < 64; ++c) mean += y[row * 64 + c];
    mean *= (1.f / 64.f);
    for (int c = 0; c < 64; ++c) {
      float d = y[row * 64 + c] - mean;
      var += d * d;
    }
    var *= (1.f / 64.f);
    float rs = rsqrtf(var + EPSF);
    size_t g = (size_t)(r0 + row) * 64;
    for (int c = 0; c < 64; ++c) {
      float val = (y[row * 64 + c] - mean) * rs * lg[c] + lb[c];
      out1f[g + c] = val;
      out1b[g + c] = f2bf(val);
    }
  }
}

// ==== x = LN(out1 + (relu(out1@(Msc+I)+sclb) @ elW + elb)) : two WMMA GEMMs ====
// mscB and elwB are transposed [N][K].
__global__ void k_ffn_ln2(const float* __restrict__ out1f, const u16* __restrict__ out1b,
                          const u16* __restrict__ mscB, const float* __restrict__ sclb,
                          const u16* __restrict__ elwB, const float* __restrict__ elb,
                          const float* __restrict__ lg, const float* __restrict__ lb,
                          float* __restrict__ x) {
  int r0 = blockIdx.x * 64;
  __shared__ __align__(16) u16 ft[64 * 64];
  __shared__ float y[64 * 64];
  int tid = threadIdx.x, wid = tid >> 5, j0 = wid * 2;
  int lane = tid & 31, n = lane & 15, moff = (lane >> 4) << 3;
  float sb = sclb[0];
#pragma unroll
  for (int j2 = 0; j2 < 2; ++j2) {
    int j = j0 + j2;
    int tm = j >> 2, tn = j & 3;
    v8f acc = vzero();
#pragma unroll
    for (int kkk = 0; kkk < 64; kkk += 32) {
      v16bf a  = load_a_rm(out1b + (size_t)(r0 + tm * 16) * 64 + kkk, 64);
      v16bf bm = load_b_tr(mscB + (size_t)(tn * 16) * 64 + kkk, 64);
      acc = WMMA_BF16(a, bm, acc);
    }
#pragma unroll
    for (int v = 0; v < 8; ++v) {
      int m = v + moff;
      ft[(tm * 16 + m) * 64 + tn * 16 + n] = f2bf(fmaxf(acc[v] + sb, 0.f));
    }
  }
  __syncthreads();
#pragma unroll
  for (int j2 = 0; j2 < 2; ++j2) {
    int j = j0 + j2;
    int tm = j >> 2, tn = j & 3;
    v8f acc = vzero();
#pragma unroll
    for (int kkk = 0; kkk < 64; kkk += 32) {
      v16bf a  = load_a_rm(ft + tm * 16 * 64 + kkk, 64);
      v16bf bm = load_b_tr(elwB + (size_t)(tn * 16) * 64 + kkk, 64);
      acc = WMMA_BF16(a, bm, acc);
    }
#pragma unroll
    for (int v = 0; v < 8; ++v) {
      int m = v + moff;
      y[(tm * 16 + m) * 64 + tn * 16 + n] =
          acc[v] + elb[tn * 16 + n] + out1f[(size_t)(r0 + tm * 16 + m) * 64 + tn * 16 + n];
    }
  }
  __syncthreads();
  if (tid < 64) {
    int row = tid;
    float mean = 0.f, var = 0.f;
    for (int c = 0; c < 64; ++c) mean += y[row * 64 + c];
    mean *= (1.f / 64.f);
    for (int c = 0; c < 64; ++c) {
      float d = y[row * 64 + c] - mean;
      var += d * d;
    }
    var *= (1.f / 64.f);
    float rs = rsqrtf(var + EPSF);
    size_t g = (size_t)(r0 + row) * 64;
    for (int c = 0; c < 64; ++c)
      x[g + c] = (y[row * 64 + c] - mean) * rs * lg[c] + lb[c];
  }
}

// ================= pooled mean -> 2-layer MLP head -> out[B] ===============
__global__ void k_head(const float* __restrict__ x, const float* __restrict__ fW1,
                       const float* __restrict__ fb1, const float* __restrict__ fW2,
                       const float* __restrict__ fb2, float* __restrict__ out) {
  int b = blockIdx.x;
  __shared__ float pooled[64];
  __shared__ float hsh[32];
  int tid = threadIdx.x;
  if (tid < 64) {
    float s = 0.f;
    const float* xb = x + (size_t)b * TT * NN + tid;
    for (int t = 0; t < TT; ++t) s += xb[(size_t)t * NN];
    pooled[tid] = s * (1.f / TT);
  }
  __syncthreads();
  if (tid < 32) {
    float hv = fb1[tid];
    for (int n2 = 0; n2 < 64; ++n2) hv += pooled[n2] * fW1[n2 * 32 + tid];
    hsh[tid] = fmaxf(hv, 0.f);
  }
  __syncthreads();
  if (tid == 0) {
    float o = fb2[0];
    for (int j = 0; j < 32; ++j) o += hsh[j] * fW2[j];
    out[b] = o;
  }
}

// ================================= host ====================================

extern "C" void kernel_launch(void* const* d_in, const int* in_sizes, int n_in,
                              void* d_out, int out_size, void* d_ws, size_t ws_size,
                              hipStream_t stream) {
  (void)in_sizes; (void)n_in; (void)out_size; (void)ws_size;
  const float* inputs = (const float*)d_in[0];
  const float* pos_emb = (const float*)d_in[1];
  const float* K3 = (const float*)d_in[2];
  const float* qw = (const float*)d_in[3];  const float* qb = (const float*)d_in[4];
  const float* kw = (const float*)d_in[5];  const float* kb = (const float*)d_in[6];
  const float* vw = (const float*)d_in[7];  const float* vb = (const float*)d_in[8];
  const float* tqW1 = (const float*)d_in[9];  const float* tqb1 = (const float*)d_in[10];
  const float* tqW2 = (const float*)d_in[11]; const float* tqb2 = (const float*)d_in[12];
  const float* tqW3 = (const float*)d_in[13]; const float* tqb3 = (const float*)d_in[14];
  const float* Wlt = (const float*)d_in[15];
  const float* ln1g = (const float*)d_in[16]; const float* ln1b = (const float*)d_in[17];
  const float* ln2g = (const float*)d_in[18]; const float* ln2b = (const float*)d_in[19];
  const float* sclw = (const float*)d_in[20]; const float* sclb = (const float*)d_in[21];
  const float* elW = (const float*)d_in[22];  const float* elb = (const float*)d_in[23];
  const float* fW1 = (const float*)d_in[24];  const float* fb1 = (const float*)d_in[25];
  const float* fW2 = (const float*)d_in[26];  const float* fb2 = (const float*)d_in[27];

  char* wsp = (char*)d_ws;
  size_t off = 0;
  auto carve = [&](size_t bytes) -> char* {
    char* p = wsp + off;
    off = (off + bytes + 255) & ~(size_t)255;
    return p;
  };
  float* x     = (float*)carve((size_t)BB * TT * NN * 4);
  float* out1f = (float*)carve((size_t)BB * TT * NN * 4);
  u16* out1b   = (u16*)carve((size_t)BB * TT * NN * 2);
  u16* xqkv    = (u16*)carve(3 * QKV_SZ * 2);                  // [3][B][H][T][N]
  u16* sft     = (u16*)carve((size_t)BB * HH * TT * TT * 2);   // 128 MB, L2-resident
  float* auv   = (float*)carve((size_t)BB * HH * TT * NN * 4);
  float* gate  = (float*)carve((size_t)BB * HH * TT * 2 * 4);
  u16* a_x     = (u16*)carve((size_t)BB * TT * HH * NN * 2);
  u16* mqkvB   = (u16*)carve((size_t)LL * 3 * HH * NN * NN * 2);  // transposed [N][K]
  u16* mscB    = (u16*)carve((size_t)LL * NN * NN * 2);           // transposed
  u16* wltB    = (u16*)carve((size_t)LL * HH * NN * NN * 2);      // transposed [64][512]
  u16* elwB    = (u16*)carve((size_t)LL * NN * NN * 2);           // transposed
  u16* w1B     = (u16*)carve((size_t)LL * HH * TT * DD1 * 2);     // transposed [64][1024]
  u16* w2B     = (u16*)carve((size_t)LL * HH * DD1 * DD1 * 2);    // transposed
  float* b3h   = (float*)carve((size_t)LL * 3 * HH * 4);          // packed qkv biases

  u16* xqw = xqkv;
  u16* xkw = xqkv + QKV_SZ;
  u16* xvw = xqkv + 2 * QKV_SZ;

  k_addpos<<<1024, 256, 0, stream>>>(inputs, pos_emb, x, BB * TT * NN);
  k_pack_bias<<<1, 64, 0, stream>>>(qb, kb, vb, b3h);
  k_build_mqkv<<<768, 256, 0, stream>>>(K3, qw, kw, vw, mqkvB);
  k_build_msc<<<32, 256, 0, stream>>>(K3, sclw, mscB);
  k_f2bf_tr<<<256, 256, 0, stream>>>(Wlt, wltB, 512, 64, LL * 512 * 64);
  k_f2bf_tr<<<32, 256, 0, stream>>>(elW, elwB, 64, 64, LL * 64 * 64);
  k_f2bf_tr<<<4096, 256, 0, stream>>>(tqW1, w1B, 1024, 64, LL * HH * 1024 * 64);
  k_f2bf_tr<<<256, 256, 0, stream>>>(tqW2, w2B, 64, 64, LL * HH * 64 * 64);

  const size_t smem_scores = (size_t)64 * 1024 * sizeof(float) + (size_t)64 * 64 * sizeof(u16);

  for (int l = 0; l < LL; ++l) {
    k_qkv<<<128, 256, 0, stream>>>(x, mqkvB + (size_t)l * 3 * HH * NN * NN,
                                   b3h + (size_t)l * 3 * HH, xqkv);
    k_scores<<<dim3(16, 64), 256, smem_scores, stream>>>(xqw, xkw, sft);
    k_auv<<<dim3(16, 64), 256, 0, stream>>>(sft, xvw, auv);
    k_gate<<<dim3(16, 64), 256, 0, stream>>>(sft, w1B + (size_t)l * HH * TT * DD1,
                                             tqb1 + l * HH * DD1,
                                             w2B + (size_t)l * HH * DD1 * DD1,
                                             tqb2 + l * HH * DD1,
                                             tqW3 + (size_t)l * HH * DD1 * 2,
                                             tqb3 + l * HH * 2, gate);
    k_mix<<<4096, 256, 0, stream>>>(gate, auv, xvw, a_x);
    k_proj_ln1<<<128, 256, 0, stream>>>(x, a_x, wltB + (size_t)l * HH * NN * NN,
                                        ln1g + l * NN, ln1b + l * NN, out1f, out1b);
    k_ffn_ln2<<<128, 256, 0, stream>>>(out1f, out1b, mscB + (size_t)l * NN * NN, sclb + l,
                                       elwB + (size_t)l * NN * NN, elb + l * NN,
                                       ln2g + l * NN, ln2b + l * NN, x);
  }
  k_head<<<BB, 256, 0, stream>>>(x, fW1, fb1, fW2, fb2, (float*)d_out);
}
